// TinySeq2Seq_48095043780612
// MI455X (gfx1250) — matrix operand order repaired and checked by reading, compiled, not run on hardware
//
#include <hip/hip_runtime.h>
#include <hip/hip_bf16.h>
#include <cstdint>

typedef __attribute__((ext_vector_type(16))) __bf16    v16bf;
typedef __attribute__((ext_vector_type(8)))  float     v8f;
typedef __attribute__((ext_vector_type(4)))  unsigned  u32x4;
typedef __attribute__((ext_vector_type(8)))  unsigned  u32x8;

// ---------------- problem constants (match reference) ----------------
static constexpr int B  = 32;
static constexpr int T  = 128;
static constexpr int E  = 512;
static constexpr int H  = 1024;
static constexpr int V  = 32000;
static constexpr int L  = 64;       // max_len
static constexpr int G4 = 4 * H;    // 4096 gate width
static constexpr int BOS = 1;

// ---------------- GEMM tiling ----------------
static constexpr int KC   = 256;        // K-chunk staged in LDS
static constexpr int SROW = KC + 8;     // +8 bf16 pad -> 528B rows, conflict-free

// =====================================================================
// Tensor Data Mover: DMA one 32 x 256 bf16 activation chunk into LDS.
// 2D descriptor (groups 0+1 only).  LDS padding replicates the SROW
// layout in hardware: 4 DWORDs (8 bf16) after every 128 DWORDs (256 bf16).
// lds_addr is derived from the real destination pointer (generic LDS
// pointers carry the wave-relative LDS byte offset in their low 32 bits),
// which also forces the LDS object to escape so the compiler cannot fold
// the subsequent ds_load reads of the DMA-written data.
// =====================================================================
__device__ __forceinline__ void tdm_issue(const __hip_bfloat16* Xp, int Kp,
                                          int kb, __hip_bfloat16* ldsDst)
{
    const unsigned long long ga =
        (unsigned long long)Xp + (unsigned long long)kb * 2ull;
    const unsigned dim0   = (unsigned)(Kp - kb);             // remaining row len
    const unsigned ldsOff = (unsigned)(uintptr_t)ldsDst;     // LDS byte offset

    u32x4 g0;
    g0.x = (unsigned)__builtin_amdgcn_readfirstlane((int)1u);        // count=1
    g0.y = (unsigned)__builtin_amdgcn_readfirstlane((int)ldsOff);    // lds_addr
    g0.z = (unsigned)__builtin_amdgcn_readfirstlane((int)(unsigned)(ga & 0xFFFFFFFFu));
    g0.w = (unsigned)__builtin_amdgcn_readfirstlane(
               (int)((unsigned)((ga >> 32) & 0x01FFFFFFu) | (2u << 30))); // type=2

    // data_size=1 (2B), pad_enable, pad_interval=6 (128 DW), pad_amount=3 (4 DW)
    u32x8 g1;
    g1[0] = (unsigned)__builtin_amdgcn_readfirstlane(
                (int)((1u << 16) | (1u << 20) | (6u << 22) | (3u << 25)));
    g1[1] = (unsigned)__builtin_amdgcn_readfirstlane((int)((dim0 & 0xFFFFu) << 16));
    g1[2] = (unsigned)__builtin_amdgcn_readfirstlane((int)((dim0 >> 16) | (32u << 16)));
    g1[3] = (unsigned)__builtin_amdgcn_readfirstlane((int)(256u << 16)); // tile_dim0
    g1[4] = (unsigned)__builtin_amdgcn_readfirstlane((int)32u);          // tile_dim1
    g1[5] = (unsigned)__builtin_amdgcn_readfirstlane((int)(unsigned)Kp); // dim0 stride
    g1[6] = (unsigned)__builtin_amdgcn_readfirstlane(0);
    g1[7] = (unsigned)__builtin_amdgcn_readfirstlane(0);

    // ldsDst passed as an operand so the pointed-to LDS is known-escaped;
    // "memory" clobber then keeps the DMA write visible to later ds_loads.
    asm volatile("tensor_load_to_lds %0, %1"
                 :: "s"(g0), "s"(g1), "v"(ldsDst)
                 : "memory");
}

// =====================================================================
// Core GEMM: D[32 x N] = [X0 | X1] (32 x (K0+K1)) @ [W0 | W1]^T + bias
//   Activations staged via TDM with LDS double-buffering; WMMA fragments
//   register-double-buffered so loads overlap the matrix pipe.
//   Grid.x = N/64, block = 256 (8 waves); wave -> one 16x16 tile.
// =====================================================================
__global__ __launch_bounds__(256) void gemm_rows32(
    const __hip_bfloat16* __restrict__ X0, int K0,
    const __hip_bfloat16* __restrict__ X1, int K1,
    const __hip_bfloat16* __restrict__ W0,
    const __hip_bfloat16* __restrict__ W1,
    const float* __restrict__ bias,
    float* __restrict__ out, long outStride)
{
    __shared__ __align__(16) __hip_bfloat16 sX[2][32 * SROW];

    const int tid   = threadIdx.x;
    const int wave  = tid >> 5;
    const int lane  = tid & 31;
    const int M0    = (wave & 1) * 16;
    const int N0    = blockIdx.x * 64 + (wave >> 1) * 16;
    const int nlo   = lane & 15;     // column within tile (B/D layout)
    const int khalf = lane >> 4;     // 0: lanes 0-15, 1: lanes 16-31

    v8f acc;
    const float bv = bias[N0 + nlo];
    #pragma unroll
    for (int v = 0; v < 8; ++v) acc[v] = bv;

    const int nchunks = (K0 + K1) / KC;

    auto params = [&](int ch, const __hip_bfloat16*& Xp,
                      const __hip_bfloat16*& Wp, int& Kp, int& kb) {
        const bool sec = (ch * KC) >= K0;
        Xp = sec ? X1 : X0;
        Wp = sec ? W1 : W0;
        Kp = sec ? K1 : K0;
        kb = ch * KC - (sec ? K0 : 0);
    };

    // prologue: DMA chunk 0 into buffer 0
    {
        const __hip_bfloat16 *Xp, *Wp; int Kp, kb;
        params(0, Xp, Wp, Kp, kb);
        if (wave == 0) {
            tdm_issue(Xp, Kp, kb, &sX[0][0]);
            __builtin_amdgcn_s_wait_tensorcnt(0);
        }
    }
    __syncthreads();

    for (int ch = 0; ch < nchunks; ++ch) {
        const int cur = ch & 1;

        // DMA next chunk into the other buffer while this one computes
        if (ch + 1 < nchunks && wave == 0) {
            const __hip_bfloat16 *Xp, *Wp; int Kp, kb;
            params(ch + 1, Xp, Wp, Kp, kb);
            tdm_issue(Xp, Kp, kb, &sX[cur ^ 1][0]);
        }

        const __hip_bfloat16 *Xp, *Wp; int Kp, kb;
        params(ch, Xp, Wp, Kp, kb);
        const __hip_bfloat16* sbase = &sX[cur][0];
        const __hip_bfloat16* wrow  =
            Wp + (size_t)(N0 + nlo) * Kp + kb + khalf * 8;

        union Frag { uint4 q[2]; v16bf v; };
        Frag Af[2], Bf[2];
        auto loadA = [&](Frag& f, int ks) {
            const __hip_bfloat16* a = &sbase[(M0 + nlo) * SROW + ks + khalf * 8];
            f.q[0] = *(const uint4*)(a);        // K = ks + khalf*8 + 0..7
            f.q[1] = *(const uint4*)(a + 16);   // K = ks + 16 + khalf*8 + 0..7
        };
        auto loadB = [&](Frag& f, int ks) {
            f.q[0] = *(const uint4*)(wrow + ks);
            f.q[1] = *(const uint4*)(wrow + ks + 16);
        };

        loadA(Af[0], 0); loadB(Bf[0], 0);
        #pragma unroll
        for (int i = 0; i < KC / 32; ++i) {
            const int p = i & 1;
            if (i + 1 < KC / 32) {              // fetch ahead under the WMMA
                loadA(Af[p ^ 1], (i + 1) * 32);
                loadB(Bf[p ^ 1], (i + 1) * 32);
            }
            acc = __builtin_amdgcn_wmma_f32_16x16x32_bf16(
                      false, Af[p].v, false, Bf[p].v, (short)0, acc, false, false);
        }

        // prefetch next chunk's weight row (global_prefetch_b8)
        if (ch + 1 < nchunks) {
            const __hip_bfloat16 *Xq, *Wq; int Kq, kb2;
            params(ch + 1, Xq, Wq, Kq, kb2);
            __builtin_prefetch(Wq + (size_t)(N0 + nlo) * Kq + kb2 + khalf * 16, 0, 1);
        }

        __syncthreads();                        // all waves done reading sX[cur]
        if (ch + 1 < nchunks) {
            if (wave == 0) __builtin_amdgcn_s_wait_tensorcnt(0);
            __syncthreads();                    // chunk ch+1 visible to all waves
        }
    }

    // D layout: VGPR v -> row M0 + v + 8*khalf, column N0 + nlo
    #pragma unroll
    for (int v = 0; v < 8; ++v) {
        const int row = M0 + v + 8 * khalf;
        out[(size_t)row * outStride + N0 + nlo] = acc[v];
    }
}

// =====================================================================
// LSTM pointwise: gates G[32 x 4096] (i|f|g|o), update c (f32), emit h (bf16)
// =====================================================================
__global__ void lstm_pointwise(const float* __restrict__ G,
                               float* __restrict__ c,
                               __hip_bfloat16* __restrict__ h)
{
    const int i = blockIdx.x * blockDim.x + threadIdx.x;
    if (i >= B * H) return;
    const int b = i >> 10;
    const int j = i & (H - 1);
    const float* g = G + (size_t)b * G4;
    const float gi = g[j];
    const float gf = g[H + j];
    const float gg = g[2 * H + j];
    const float go = g[3 * H + j];
    const float si = 1.0f / (1.0f + expf(-gi));
    const float sf = 1.0f / (1.0f + expf(-gf));
    const float so = 1.0f / (1.0f + expf(-go));
    const float c2 = sf * c[i] + si * tanhf(gg);
    c[i] = c2;
    h[i] = __float2bfloat16(so * tanhf(c2));
}

// =====================================================================
// helpers: f32 -> bf16 convert, zero fill, embedding gathers, argmax
// =====================================================================
__global__ void f32_to_bf16(const float* __restrict__ in,
                            __hip_bfloat16* __restrict__ out, int n)
{
    const int i = blockIdx.x * blockDim.x + threadIdx.x;
    if (i < n) out[i] = __float2bfloat16(in[i]);
}

__global__ void zero_u32(unsigned* __restrict__ p, int n)
{
    const int i = blockIdx.x * blockDim.x + threadIdx.x;
    if (i < n) p[i] = 0u;
}

// xs[t][b][e] = bf16(emb[src[b][t]][e])
__global__ void gather_enc(const float* __restrict__ emb,
                           const int* __restrict__ src,
                           __hip_bfloat16* __restrict__ xs)
{
    const long i = (long)blockIdx.x * blockDim.x + threadIdx.x;
    if (i >= (long)T * B * E) return;
    const int  e = (int)(i & (E - 1));
    const long r = i >> 9;
    const int  b = (int)(r & (B - 1));
    const int  t = (int)(r >> 5);
    const long tok = src[b * T + t];
    xs[i] = __float2bfloat16(emb[tok * E + e]);
}

// x[b][e] = bf16(emb[tok[b]][e]);  tok = BOS on first step
__global__ void gather_dec(const float* __restrict__ emb,
                           const int* __restrict__ tok, int useBos,
                           __hip_bfloat16* __restrict__ x)
{
    const int i = blockIdx.x * blockDim.x + threadIdx.x;
    if (i >= B * E) return;
    const int b = i >> 9;
    const int e = i & (E - 1);
    const long t = useBos ? BOS : tok[b];
    x[i] = __float2bfloat16(emb[t * E + e]);
}

// first-occurrence argmax over each batch row of the step's logits
__global__ __launch_bounds__(256) void argmax_rows(
    const float* __restrict__ logits, long rowStride, int* __restrict__ tok)
{
    __shared__ float sv[256];
    __shared__ int   si[256];
    const int b   = blockIdx.x;
    const int tid = threadIdx.x;
    const float* row = logits + (size_t)b * rowStride;
    float best = -__builtin_inff();
    int   bi   = 0;
    for (int j = tid; j < V; j += 256) {
        const float v = row[j];
        if (v > best) { best = v; bi = j; }
    }
    sv[tid] = best; si[tid] = bi;
    __syncthreads();
    for (int s = 128; s > 0; s >>= 1) {
        if (tid < s) {
            if (sv[tid + s] > sv[tid] ||
                (sv[tid + s] == sv[tid] && si[tid + s] < si[tid])) {
                sv[tid] = sv[tid + s];
                si[tid] = si[tid + s];
            }
        }
        __syncthreads();
    }
    if (tid == 0) tok[b] = si[0];
}

// =====================================================================
// host orchestration
// =====================================================================
extern "C" void kernel_launch(void* const* d_in, const int* in_sizes, int n_in,
                              void* d_out, int out_size, void* d_ws, size_t ws_size,
                              hipStream_t stream)
{
    (void)in_sizes; (void)n_in; (void)out_size; (void)ws_size;
    typedef __hip_bfloat16 bf;

    const int*   src    = (const int*)  d_in[0];
    const float* emb    = (const float*)d_in[1];
    const float* eW0ih  = (const float*)d_in[2];
    const float* eW0hh  = (const float*)d_in[3];
    const float* eb0    = (const float*)d_in[4];
    const float* eW1ih  = (const float*)d_in[5];
    const float* eW1hh  = (const float*)d_in[6];
    const float* eb1    = (const float*)d_in[7];
    const float* dW0ih  = (const float*)d_in[8];
    const float* dW0hh  = (const float*)d_in[9];
    const float* db0    = (const float*)d_in[10];
    const float* dW1ih  = (const float*)d_in[11];
    const float* dW1hh  = (const float*)d_in[12];
    const float* db1    = (const float*)d_in[13];
    const float* outW   = (const float*)d_in[14];
    const float* outB   = (const float*)d_in[15];
    float* out = (float*)d_out;                    // [B, L, V]

    char* basep = (char*)d_ws;
    size_t off = 0;
    auto carve = [&](size_t bytes) -> char* {
        char* p = basep + off;
        off += (bytes + 255) & ~(size_t)255;
        return p;
    };
    bf* wE0ih = (bf*)carve((size_t)G4 * E * 2);
    bf* wE0hh = (bf*)carve((size_t)G4 * H * 2);
    bf* wE1ih = (bf*)carve((size_t)G4 * H * 2);
    bf* wE1hh = (bf*)carve((size_t)G4 * H * 2);
    bf* wD0ih = (bf*)carve((size_t)G4 * E * 2);
    bf* wD0hh = (bf*)carve((size_t)G4 * H * 2);
    bf* wD1ih = (bf*)carve((size_t)G4 * H * 2);
    bf* wD1hh = (bf*)carve((size_t)G4 * H * 2);
    bf* wOut  = (bf*)carve((size_t)V * H * 2);
    bf* xs    = (bf*)carve((size_t)T * B * E * 2);
    bf* hs0   = (bf*)carve((size_t)T * B * H * 2);
    bf* h1buf = (bf*)carve((size_t)B * H * 2);
    bf* dh0   = (bf*)carve((size_t)B * H * 2);
    bf* dh1   = (bf*)carve((size_t)B * H * 2);
    bf* zbuf  = (bf*)carve((size_t)B * H * 2);
    bf* xdec  = (bf*)carve((size_t)B * E * 2);
    float* c0 = (float*)carve((size_t)B * H * 4);
    float* c1 = (float*)carve((size_t)B * H * 4);
    float* G  = (float*)carve((size_t)B * G4 * 4);
    int* tok  = (int*)carve((size_t)B * 4);

    auto cvt = [&](const float* s, bf* d, size_t n) {
        f32_to_bf16<<<(unsigned)((n + 255) / 256), 256, 0, stream>>>(s, d, (int)n);
    };

    // state init (buffers are not re-poisoned between replays)
    zero_u32<<<(B*H*2/4 + 255)/256, 256, 0, stream>>>((unsigned*)zbuf, B*H*2/4);
    zero_u32<<<(B*H   + 255)/256, 256, 0, stream>>>((unsigned*)c0, B*H);
    zero_u32<<<(B*H   + 255)/256, 256, 0, stream>>>((unsigned*)c1, B*H);

    // weight conversion to bf16 (~122 MB total -> stays resident in 192MB L2)
    cvt(eW0ih, wE0ih, (size_t)G4 * E);
    cvt(eW0hh, wE0hh, (size_t)G4 * H);
    cvt(eW1ih, wE1ih, (size_t)G4 * H);
    cvt(eW1hh, wE1hh, (size_t)G4 * H);
    cvt(dW0ih, wD0ih, (size_t)G4 * E);
    cvt(dW0hh, wD0hh, (size_t)G4 * H);
    cvt(dW1ih, wD1ih, (size_t)G4 * H);
    cvt(dW1hh, wD1hh, (size_t)G4 * H);
    cvt(outW,  wOut,  (size_t)V  * H);

    {
        const long n = (long)T * B * E;
        gather_enc<<<(unsigned)((n + 255) / 256), 256, 0, stream>>>(emb, src, xs);
    }

    const unsigned pwGrid = (B * H + 255) / 256;

    // encoder layer 0 (h_t -> hs0[t], doubles as recurrence input)
    for (int t = 0; t < T; ++t) {
        const bf* hprev = (t == 0) ? zbuf : hs0 + (size_t)(t - 1) * B * H;
        gemm_rows32<<<G4 / 64, 256, 0, stream>>>(
            xs + (size_t)t * B * E, E, hprev, H, wE0ih, wE0hh, eb0, G, G4);
        lstm_pointwise<<<pwGrid, 256, 0, stream>>>(G, c0, hs0 + (size_t)t * B * H);
    }
    // encoder layer 1 (final h1buf/c1 = decoder layer-1 init)
    for (int t = 0; t < T; ++t) {
        const bf* hprev = (t == 0) ? zbuf : h1buf;
        gemm_rows32<<<G4 / 64, 256, 0, stream>>>(
            hs0 + (size_t)t * B * H, H, hprev, H, wE1ih, wE1hh, eb1, G, G4);
        lstm_pointwise<<<pwGrid, 256, 0, stream>>>(G, c1, h1buf);
    }

    // greedy autoregressive decode
    const long outRow = (long)L * V;
    for (int t = 0; t < L; ++t) {
        gather_dec<<<(B * E + 255) / 256, 256, 0, stream>>>(emb, tok, t == 0 ? 1 : 0, xdec);

        const bf* h0prev = (t == 0) ? hs0 + (size_t)(T - 1) * B * H : dh0;
        gemm_rows32<<<G4 / 64, 256, 0, stream>>>(
            xdec, E, h0prev, H, wD0ih, wD0hh, db0, G, G4);
        lstm_pointwise<<<pwGrid, 256, 0, stream>>>(G, c0, dh0);

        const bf* h1prev = (t == 0) ? h1buf : dh1;
        gemm_rows32<<<G4 / 64, 256, 0, stream>>>(
            dh0, H, h1prev, H, wD1ih, wD1hh, db1, G, G4);
        lstm_pointwise<<<pwGrid, 256, 0, stream>>>(G, c1, dh1);

        gemm_rows32<<<V / 64, 256, 0, stream>>>(
            dh1, H, (const bf*)nullptr, 0, wOut, (const bf*)nullptr, outB,
            out + (size_t)t * V, outRow);

        argmax_rows<<<B, 256, 0, stream>>>(out + (size_t)t * V, outRow, tok);
    }
}
